// Net_88965952570002
// MI455X (gfx1250) — matrix operand order repaired and checked by reading, not compiled, not run on hardware
//
#include <hip/hip_runtime.h>
#include <math.h>

// ---------------------------------------------------------------------------
// LSTM captioner for MI455X (gfx1250, wave32, WMMA).
//
// GEMMs: Y[64,N] = X[64,512] @ W[N,512]^T via v_wmma_f32_16x16x32_bf16
// (bf16 operands, fp32 accum). bf16 weights (49 MB) are L2-resident.
// Activation tiles are staged to LDS once per block with the CDNA5 async
// copy path (global_load_async_to_lds_b128 + s_wait_asynccnt) and consumed
// by WMMA A-fragments through ds_load_b128, removing redundant global reads.
// Dead work removed analytically:
//   * encoder cell2 input = concat([0, h1])  -> only w_ih2[:, H:2H] used
//   * decoder cell1 input = 0                -> w_ihd1 GEMM removed (bias only)
// ---------------------------------------------------------------------------

#define B_  64
#define T_  48
#define H_  512
#define V_  32000
#define G4  (4 * H_)   // 2048

typedef __attribute__((ext_vector_type(16))) __bf16 v16bf;
typedef __attribute__((ext_vector_type(8)))  __bf16 v8bf;
typedef __attribute__((ext_vector_type(8)))  float  v8f;

// ---- async global -> LDS staging of one 16x512 bf16 tile ------------------
// 1024 chunks of 16B; 128 threads -> 8 async b128 transfers per thread.
// Tracked by ASYNCcnt; caller must s_wait_asynccnt + barrier before reading.
__device__ __forceinline__ void stage_tile_async(__bf16* dst /*LDS 16x512*/,
                                                 const __bf16* X, int ld,
                                                 int m0) {
#pragma unroll
  for (int it = 0; it < 8; ++it) {
    const int c   = it * 128 + threadIdx.x;     // 0..1023
    const int row = c >> 6;                     // 0..15
    const int col = (c & 63) << 3;              // element 0..504 step 8
    const __bf16* g = X + (m0 + row) * ld + col;
    // low 32 bits of a shared->generic pointer == LDS byte offset
    const unsigned la = (unsigned)(size_t)(dst + row * H_ + col);
    asm volatile("global_load_async_to_lds_b128 %0, %1, off"
                 :: "v"(la), "v"(g) : "memory");
  }
}
__device__ __forceinline__ void wait_async_and_sync() {
  asm volatile("s_wait_asynccnt 0x0" ::: "memory");
  __syncthreads();
}

// ---- fragment loaders -----------------------------------------------------
// A (16x32 bf16) per ISA layout:
//   lanes 0-15 : row lane,    K = {k0+0..7, k0+16..23}
//   lanes 16-31: row lane-16, K = {k0+8..15, k0+24..31}
// src is the LDS tile (row stride 512) -> lowers to ds_load_b128 pairs.
__device__ __forceinline__ v16bf load_frag_a(const __bf16* X, int k0,
                                             int lane) {
  const int row  = lane & 15;
  const int koff = (lane >> 4) << 3;            // 0 or 8
  const __bf16* p = X + row * H_ + k0 + koff;
  v8bf lo = *(const v8bf*)(p);
  v8bf hi = *(const v8bf*)(p + 16);
  v16bf r;
#pragma unroll
  for (int i = 0; i < 8; ++i) { r[i] = lo[i]; r[i + 8] = hi[i]; }
  return r;
}

// B (32x16 bf16) from row-major W[N][512]: B[k][n] = W[nbase+n][k].
// lane (mod 16) selects output column n; K = {k0+8h+0..7, k0+8h+16..23}, h = lane/16.
__device__ __forceinline__ v16bf load_frag_b(const __bf16* W, int nbase,
                                             int k0, int lane) {
  const int row  = nbase + (lane & 15);
  const int koff = (lane >> 4) << 3;            // 0 or 8
  const __bf16* p = W + row * H_ + k0 + koff;
  v8bf lo = *(const v8bf*)(p);
  v8bf hi = *(const v8bf*)(p + 16);
  v16bf r;
#pragma unroll
  for (int i = 0; i < 8; ++i) { r[i] = lo[i]; r[i + 8] = hi[i]; }
  return r;
}

#define WMMA_BF16(a, b, c) __builtin_amdgcn_wmma_f32_16x16x32_bf16( \
    false, (a), false, (b), (short)0, (c), false, false)

// K=512 dot, A from LDS tile: 16 x v_wmma_f32_16x16x32_bf16
__device__ __forceinline__ v8f gemm_k512(const __bf16* a_lds,
                                         const __bf16* W, int nbase,
                                         int lane, v8f acc) {
#pragma unroll 4
  for (int k0 = 0; k0 < H_; k0 += 32) {
    v16bf a = load_frag_a(a_lds, k0, lane);
    v16bf b = load_frag_b(W, nbase, k0, lane);
    acc = WMMA_BF16(a, b, acc);
  }
  return acc;
}

__device__ __forceinline__ float sigm(float x) { return 1.f / (1.f + expf(-x)); }

// ---- fused LSTM cell: up to 3 GEMM inputs, bias, gate nonlinearity --------
// grid = (H/16, B/16), block = 128 (4 waves; wave g owns gate block g).
__global__ __launch_bounds__(128) void lstm_cell_kernel(
    const __bf16* X1, int ld1, const __bf16* W1,
    const __bf16* X2, int ld2, const __bf16* W2,
    const __bf16* X3, int ld3, const __bf16* W3,
    int nX, const float* bias,
    const float* c_in, float* c_out, __bf16* h_out) {
  __shared__ alignas(16) __bf16 xt[3][16][H_];   // staged activations (48KB)
  __shared__ float gl[4][16][16];                // gate exchange (4KB)
  const int lane = threadIdx.x & 31;
  const int g    = threadIdx.x >> 5;            // gate: 0=i 1=f 2=g 3=o
  const int m0   = blockIdx.y << 4;
  const int n0   = blockIdx.x << 4;
  const int nrow = g * H_ + n0;                 // row base in W / bias

  stage_tile_async(&xt[0][0][0], X1, ld1, m0);
  if (nX > 1) stage_tile_async(&xt[1][0][0], X2, ld2, m0);
  if (nX > 2) stage_tile_async(&xt[2][0][0], X3, ld3, m0);
  wait_async_and_sync();

  v8f acc = {0.f, 0.f, 0.f, 0.f, 0.f, 0.f, 0.f, 0.f};
  acc = gemm_k512(&xt[0][0][0], W1, nrow, lane, acc);
  if (nX > 1) acc = gemm_k512(&xt[1][0][0], W2, nrow, lane, acc);
  if (nX > 2) acc = gemm_k512(&xt[2][0][0], W3, nrow, lane, acc);

  // D layout: lanes 0-15 -> rows r, col lane; lanes 16-31 -> rows 8+r.
  const float bv  = bias[nrow + (lane & 15)];
  const int mbase = (lane >> 4) << 3;
#pragma unroll
  for (int r = 0; r < 8; ++r) gl[g][mbase + r][lane & 15] = acc[r] + bv;
  __syncthreads();

#pragma unroll
  for (int e = threadIdx.x; e < 256; e += 128) {
    const int m = e >> 4, n = e & 15;
    const int idx = (m0 + m) * H_ + (n0 + n);
    const float gi = gl[0][m][n], gf = gl[1][m][n];
    const float gg = gl[2][m][n], go = gl[3][m][n];
    const float cn = sigm(gf) * c_in[idx] + sigm(gi) * tanhf(gg);
    c_out[idx] = cn;
    h_out[idx] = (__bf16)(sigm(go) * tanhf(cn));
  }
}

// ---- classifier GEMM: logits[64,32000] = h2 @ w_o^T + b_o -----------------
// grid = (V/256, B/16), block = 128. Block stages one h2 tile to LDS; each
// wave computes an N=64 strip: 1 LDS A-fragment feeds 4 WMMAs per K-step.
__global__ __launch_bounds__(128) void logits_kernel(
    const __bf16* h2, const __bf16* wo, const float* bo, float* out) {
  __shared__ alignas(16) __bf16 xt[16][H_];      // 16KB
  const int lane = threadIdx.x & 31;
  const int w    = threadIdx.x >> 5;
  const int m0   = blockIdx.y << 4;
  const int nb   = (blockIdx.x << 8) + (w << 6); // wave's N strip base

  stage_tile_async(&xt[0][0], h2, H_, m0);
  wait_async_and_sync();

  v8f acc[4];
#pragma unroll
  for (int t = 0; t < 4; ++t)
    acc[t] = (v8f){0.f, 0.f, 0.f, 0.f, 0.f, 0.f, 0.f, 0.f};

#pragma unroll 4
  for (int k0 = 0; k0 < H_; k0 += 32) {
    const v16bf a = load_frag_a(&xt[0][0], k0, lane);
#pragma unroll
    for (int t = 0; t < 4; ++t) {
      const v16bf b = load_frag_b(wo, nb + (t << 4), k0, lane);
      acc[t] = WMMA_BF16(a, b, acc[t]);
    }
  }

  const int mb = m0 + ((lane >> 4) << 3);
#pragma unroll
  for (int t = 0; t < 4; ++t) {
    const int   n  = nb + (t << 4) + (lane & 15);
    const float bv = bo[n];
#pragma unroll
    for (int r = 0; r < 8; ++r) out[(mb + r) * V_ + n] = acc[t][r] + bv;
  }
}

// ---- per-row logsumexp + CE (one block per batch row, fixed-order tree) ---
__global__ __launch_bounds__(256) void ce_kernel(const float* logits,
                                                 const int* ids, int tstep,
                                                 float* ce_out) {
  const int b = blockIdx.x;
  const float* row = logits + b * V_;
  float m = -3.0e38f, s = 0.f;
  for (int j = threadIdx.x; j < V_; j += 256) {
    const float x = row[j];
    if (x > m) { s = s * expf(m - x) + 1.f; m = x; }
    else       { s += expf(x - m); }
  }
  __shared__ float sm[256], ss[256];
  sm[threadIdx.x] = m; ss[threadIdx.x] = s;
  __syncthreads();
  for (int off = 128; off > 0; off >>= 1) {
    if (threadIdx.x < off) {
      const float m2 = sm[threadIdx.x + off], s2 = ss[threadIdx.x + off];
      const float M  = fmaxf(sm[threadIdx.x], m2);
      ss[threadIdx.x] = ss[threadIdx.x] * expf(sm[threadIdx.x] - M)
                      + s2 * expf(m2 - M);
      sm[threadIdx.x] = M;
    }
    __syncthreads();
  }
  if (threadIdx.x == 0) {
    const int tgt = ids[b * T_ + tstep + 1];
    ce_out[b] = (sm[0] + logf(ss[0])) - row[tgt];
  }
}

__global__ void final_kernel(const float* ce, int n, float* out) {
  if (threadIdx.x == 0 && blockIdx.x == 0) {
    float s = 0.f;
    for (int i = 0; i < n; ++i) s += ce[i];
    out[0] = s / (float)(B_ * B_);   // sum over steps of batch-mean, / B
  }
}

// ---- prepass helpers ------------------------------------------------------
__global__ void cvt_kernel(const float* src, __bf16* dst, int N, int Ksrc,
                           int ofs, int Kdst) {
  const int i = blockIdx.x * blockDim.x + threadIdx.x;
  if (i >= N * Kdst) return;
  const int n = i / Kdst, k = i - n * Kdst;
  dst[i] = (__bf16)src[n * Ksrc + ofs + k];
}

__global__ void bias_kernel(const float* a, const float* b, float* dst, int n) {
  const int i = blockIdx.x * blockDim.x + threadIdx.x;
  if (i < n) dst[i] = a[i] + b[i];
}

__global__ void zero_f32(float* p, int n) {
  const int i = blockIdx.x * blockDim.x + threadIdx.x;
  if (i < n) p[i] = 0.f;
}
__global__ void zero_bf16(__bf16* p, int n) {
  const int i = blockIdx.x * blockDim.x + threadIdx.x;
  if (i < n) p[i] = (__bf16)0.f;
}

// ---------------------------------------------------------------------------
extern "C" void kernel_launch(void* const* d_in, const int* in_sizes, int n_in,
                              void* d_out, int out_size, void* d_ws,
                              size_t ws_size, hipStream_t stream) {
  (void)in_sizes; (void)n_in; (void)out_size; (void)ws_size;
  const float* feat   = (const float*)d_in[0];
  const float* cap    = (const float*)d_in[1];
  const int*   ids    = (const int*)  d_in[2];
  const float* w_ih1  = (const float*)d_in[3];
  const float* w_hh1  = (const float*)d_in[4];
  const float* b_ih1  = (const float*)d_in[5];
  const float* b_hh1  = (const float*)d_in[6];
  const float* w_ih2  = (const float*)d_in[7];
  const float* w_hh2  = (const float*)d_in[8];
  const float* b_ih2  = (const float*)d_in[9];
  const float* b_hh2  = (const float*)d_in[10];
  /* w_ihd1 = d_in[11] unused: its input is identically zero */
  const float* w_hhd1 = (const float*)d_in[12];
  const float* b_ihd1 = (const float*)d_in[13];
  const float* b_hhd1 = (const float*)d_in[14];
  const float* w_ihd2 = (const float*)d_in[15];
  const float* w_hhd2 = (const float*)d_in[16];
  const float* b_ihd2 = (const float*)d_in[17];
  const float* b_hhd2 = (const float*)d_in[18];
  const float* w_o    = (const float*)d_in[19];
  const float* b_o    = (const float*)d_in[20];

  // ---- workspace layout (bytes) ----
  char* ws = (char*)d_ws;
  const size_t SW = (size_t)G4 * H_ * 2;        // 2 MB per 2048x512 bf16 mat
  size_t off = 0;
  __bf16* wih1   = (__bf16*)(ws + off); off += SW;
  __bf16* whh1   = (__bf16*)(ws + off); off += SW;
  __bf16* wih2h  = (__bf16*)(ws + off); off += SW;   // w_ih2[:, H:2H]
  __bf16* whh2   = (__bf16*)(ws + off); off += SW;
  __bf16* whhd1  = (__bf16*)(ws + off); off += SW;
  __bf16* wihd2a = (__bf16*)(ws + off); off += SW;   // w_ihd2[:, 0:H]
  __bf16* wihd2b = (__bf16*)(ws + off); off += SW;   // w_ihd2[:, H:2H]
  __bf16* whhd2  = (__bf16*)(ws + off); off += SW;
  __bf16* wo     = (__bf16*)(ws + off); off += (size_t)V_ * H_ * 2;
  __bf16* featb  = (__bf16*)(ws + off); off += (size_t)B_ * T_ * H_ * 2;
  __bf16* capb   = (__bf16*)(ws + off); off += (size_t)B_ * T_ * H_ * 2;
  float*  bias1  = (float*)(ws + off);  off += G4 * 4;
  float*  bias2  = (float*)(ws + off);  off += G4 * 4;
  float*  biasd1 = (float*)(ws + off);  off += G4 * 4;
  float*  biasd2 = (float*)(ws + off);  off += G4 * 4;
  __bf16* h1a    = (__bf16*)(ws + off); off += (size_t)B_ * H_ * 2;
  __bf16* h1b    = (__bf16*)(ws + off); off += (size_t)B_ * H_ * 2;
  __bf16* h2a    = (__bf16*)(ws + off); off += (size_t)B_ * H_ * 2;
  __bf16* h2b    = (__bf16*)(ws + off); off += (size_t)B_ * H_ * 2;
  float*  c1     = (float*)(ws + off);  off += (size_t)B_ * H_ * 4;
  float*  c2     = (float*)(ws + off);  off += (size_t)B_ * H_ * 4;
  float*  logits = (float*)(ws + off);  off += (size_t)B_ * V_ * 4;
  float*  ce_buf = (float*)(ws + off);  off += (size_t)(T_ - 1) * B_ * 4;

  auto cdiv = [](int a, int b) { return (a + b - 1) / b; };

  // ---- prepass: bf16 weight/activation conversion, bias fusion, state=0 ---
  const int NK = G4 * H_;
  cvt_kernel<<<cdiv(NK,256),256,0,stream>>>(w_ih1, wih1, G4, H_, 0, H_);
  cvt_kernel<<<cdiv(NK,256),256,0,stream>>>(w_hh1, whh1, G4, H_, 0, H_);
  cvt_kernel<<<cdiv(NK,256),256,0,stream>>>(w_ih2, wih2h, G4, 2*H_, H_, H_);
  cvt_kernel<<<cdiv(NK,256),256,0,stream>>>(w_hh2, whh2, G4, H_, 0, H_);
  cvt_kernel<<<cdiv(NK,256),256,0,stream>>>(w_hhd1, whhd1, G4, H_, 0, H_);
  cvt_kernel<<<cdiv(NK,256),256,0,stream>>>(w_ihd2, wihd2a, G4, 2*H_, 0, H_);
  cvt_kernel<<<cdiv(NK,256),256,0,stream>>>(w_ihd2, wihd2b, G4, 2*H_, H_, H_);
  cvt_kernel<<<cdiv(NK,256),256,0,stream>>>(w_hhd2, whhd2, G4, H_, 0, H_);
  cvt_kernel<<<cdiv(V_*H_,256),256,0,stream>>>(w_o, wo, V_, H_, 0, H_);
  cvt_kernel<<<cdiv(B_*T_*H_,256),256,0,stream>>>(feat, featb, B_*T_, H_, 0, H_);
  cvt_kernel<<<cdiv(B_*T_*H_,256),256,0,stream>>>(cap,  capb,  B_*T_, H_, 0, H_);
  bias_kernel<<<cdiv(G4,256),256,0,stream>>>(b_ih1,  b_hh1,  bias1,  G4);
  bias_kernel<<<cdiv(G4,256),256,0,stream>>>(b_ih2,  b_hh2,  bias2,  G4);
  bias_kernel<<<cdiv(G4,256),256,0,stream>>>(b_ihd1, b_hhd1, biasd1, G4);
  bias_kernel<<<cdiv(G4,256),256,0,stream>>>(b_ihd2, b_hhd2, biasd2, G4);
  zero_bf16<<<cdiv(B_*H_,256),256,0,stream>>>(h1a, B_*H_);
  zero_bf16<<<cdiv(B_*H_,256),256,0,stream>>>(h2a, B_*H_);
  zero_f32 <<<cdiv(B_*H_,256),256,0,stream>>>(c1, B_*H_);
  zero_f32 <<<cdiv(B_*H_,256),256,0,stream>>>(c2, B_*H_);

  const dim3 cgrid(H_ / 16, B_ / 16);   // (32,4)
  const dim3 lgrid(V_ / 256, B_ / 16);  // (125,4)
  const int  ldT = T_ * H_;             // feat/caption batch-row stride

  __bf16 *h1_in = h1a, *h1_out = h1b, *h2_in = h2a, *h2_out = h2b;

  // ---- encoder: 48 sequential steps ----
  for (int t = 0; t < T_; ++t) {
    lstm_cell_kernel<<<cgrid,128,0,stream>>>(
        featb + t * H_, ldT, wih1,  h1_in, H_, whh1,
        nullptr, 0, nullptr, 2, bias1, c1, c1, h1_out);
    lstm_cell_kernel<<<cgrid,128,0,stream>>>(
        h1_out, H_, wih2h,  h2_in, H_, whh2,
        nullptr, 0, nullptr, 2, bias2, c2, c2, h2_out);
    __bf16* t1 = h1_in; h1_in = h1_out; h1_out = t1;
    __bf16* t2 = h2_in; h2_in = h2_out; h2_out = t2;
  }

  // ---- decoder: 47 sequential steps ----
  for (int s = 0; s < T_ - 1; ++s) {
    // cell d1: input is zero -> single recurrent GEMM, fused biases
    lstm_cell_kernel<<<cgrid,128,0,stream>>>(
        h1_in, H_, whhd1,  nullptr, 0, nullptr,
        nullptr, 0, nullptr, 1, biasd1, c1, c1, h1_out);
    // cell d2: concat([cap_s, h1_new]) input + recurrent
    lstm_cell_kernel<<<cgrid,128,0,stream>>>(
        capb + s * H_, ldT, wihd2a,  h1_out, H_, wihd2b,
        h2_in, H_, whhd2, 3, biasd2, c2, c2, h2_out);
    logits_kernel<<<lgrid,128,0,stream>>>(h2_out, wo, b_o, logits);
    ce_kernel<<<B_,256,0,stream>>>(logits, ids, s, ce_buf + s * B_);
    __bf16* t1 = h1_in; h1_in = h1_out; h1_out = t1;
    __bf16* t2 = h2_in; h2_in = h2_out; h2_out = t2;
  }

  final_kernel<<<1,32,0,stream>>>(ce_buf, (T_ - 1) * B_, (float*)d_out);
}
